// DGCNNBinaryClass_52965536694407
// MI455X (gfx1250) — compile-verified, matrix operand order checked by reading, not run on hardware
//
#include <hip/hip_runtime.h>

typedef __attribute__((ext_vector_type(16))) _Float16 v16h;
typedef __attribute__((ext_vector_type(8)))  _Float16 v8h;
typedef __attribute__((ext_vector_type(8)))  float    v8f;

#define EPSV   1e-5f
#define SLOPE  0.2f
#define BATCH  16
#define NPTS   1024
#define KNN    20
#define FC1_K  (NPTS*256)
#define FC1_N  512
#define KSPLIT 16

__global__ void k_fill(float* p, float v, int n) {
  int i = blockIdx.x*blockDim.x + threadIdx.x;
  if (i < n) p[i] = v;
}

// (B,N,3) -> (B,3,N)
__global__ void k_transpose_x(const float* __restrict__ x, float* __restrict__ out) {
  int i = blockIdx.x*blockDim.x + threadIdx.x;
  if (i >= BATCH*NPTS*3) return;
  int c = i % 3, n = (i/3) % NPTS, b = i/(3*NPTS);
  out[(b*3 + c)*NPTS + n] = x[i];
}

// zero-padded fp32 -> f16 weight conversion (once per launch; L2-resident after)
__global__ void k_prep_w(const float* __restrict__ W, _Float16* __restrict__ Wp,
                         int C_out, int C2, int KPAD) {
  int i = blockIdx.x*blockDim.x + threadIdx.x;
  if (i >= C_out*KPAD) return;
  int r = i / KPAD, c = i % KPAD;
  Wp[i] = (c < C2) ? (_Float16)W[r*C2 + c] : (_Float16)0.f;
}

__global__ void k_sqnorm(const float* __restrict__ x, float* __restrict__ xx, int C) {
  int i = blockIdx.x*blockDim.x + threadIdx.x;
  if (i >= BATCH*NPTS) return;
  int b = i >> 10, n = i & (NPTS-1);
  const float* xb = x + (size_t)b*C*NPTS + n;
  float s = 0.f;
  for (int c = 0; c < C; ++c) { float v = xb[c*NPTS]; s += v*v; }
  xx[i] = s;
}

// one wave32 per point; lane caches 32 distances in registers (unrolled),
// 20 rounds of masked wave-argmax with lowest-index tie break.
__global__ void k_knn(const float* __restrict__ x, const float* __restrict__ xx,
                      int* __restrict__ idx, int C) {
  int wave = threadIdx.x >> 5;
  int lane = threadIdx.x & 31;
  int p = blockIdx.x * 8 + wave;
  int b = p >> 10, n = p & (NPTS-1);
  const float* xb = x + (size_t)b*C*NPTS;
  float d[32];
  #pragma unroll
  for (int j = 0; j < 32; ++j) d[j] = 0.f;
  for (int c = 0; c < C; ++c) {
    const float* row = xb + c*NPTS;
    float xnc = row[n];
    #pragma unroll
    for (int j = 0; j < 32; ++j) d[j] += xnc * row[j*32 + lane];
  }
  float xxn = xx[b*NPTS + n];
  #pragma unroll
  for (int j = 0; j < 32; ++j)
    d[j] = 2.f*d[j] - xxn - xx[b*NPTS + j*32 + lane];

  unsigned mask = 0;
  for (int t = 0; t < KNN; ++t) {
    float best = -3.4e38f; int bj = -1;
    #pragma unroll
    for (int j = 0; j < 32; ++j)
      if (!((mask >> j) & 1u) && d[j] > best) { best = d[j]; bj = j; }
    float wm = best;
    for (int off = 16; off > 0; off >>= 1) {
      float o = __shfl_xor(wm, off, 32);
      wm = o > wm ? o : wm;
    }
    int cand = (bj >= 0 && best == wm) ? (bj*32 + lane) : 0x7fffffff;
    for (int off = 16; off > 0; off >>= 1) {
      int o = __shfl_xor(cand, off, 32);
      cand = o < cand ? o : cand;
    }
    if ((cand & 31) == lane) mask |= 1u << (cand >> 5);
    if (lane == 0) idx[(b*NPTS + n)*KNN + t] = cand;
  }
}

// Fused EdgeConv tile kernel (weights pre-padded f16, features staged in LDS in
// B-fragment order so each lane issues one contiguous 32B v16h LDS load).
// pass 0: WMMA GEMM -> per-channel sum/sumsq (wave shuffle reduce + atomics).
// pass 1: WMMA GEMM -> folded BN + LeakyReLU -> atomicMax over k into (B,Cout,N).
__global__ void k_edgeconv(const float* __restrict__ xin, const int* __restrict__ idx,
                           const _Float16* __restrict__ Wp,
                           float* __restrict__ sum, float* __restrict__ sumsq,
                           const float* __restrict__ Asc, const float* __restrict__ Bsh,
                           float* __restrict__ xout,
                           int C_in, int C_out, int KPAD, int pass) {
  const int tiles_per_b = (NPTS*KNN)/16;     // 1280
  int tile = blockIdx.x % tiles_per_b;
  int b    = blockIdx.x / tiles_per_b;
  int s0   = tile * 16;
  int C2   = 2*C_in;
  int nchunks = KPAD >> 5;

  // B-fragment-ordered staging: sfB[chunk*512 + lane*16 + slot]
  __shared__ __align__(32) _Float16 sfB[8*512];  // max 8 chunks -> 8KB
  for (int i = threadIdx.x; i < nchunks*512; i += blockDim.x) sfB[i] = (_Float16)0.f;
  __syncthreads();

  const float* xb = xin + (size_t)b*C_in*NPTS;
  for (int i = threadIdx.x; i < C2*16; i += blockDim.x) {
    int c = i >> 4, sl = i & 15;
    int s = s0 + sl;
    int n = s / KNN, kk = s % KNN;
    float v;
    if (c < C_in) {
      int m = idx[(b*NPTS + n)*KNN + kk];
      v = xb[c*NPTS + m] - xb[c*NPTS + n];   // nbr - center
    } else {
      v = xb[(c - C_in)*NPTS + n];           // center
    }
    int ck = c & 31;
    int hh = (ck >> 3) & 1;
    int slot = (ck & 7) | ((ck >> 4) << 3);
    sfB[(c >> 5)*512 + ((sl + (hh << 4)) << 4) + slot] = (_Float16)v;
  }
  __syncthreads();

  int wave = threadIdx.x >> 5, lane = threadIdx.x & 31;
  int col = lane & 15, h = lane >> 4;
  int tiles_o = C_out >> 4;
  for (int ot = wave; ot < tiles_o; ot += 8) {
    v8f acc = {};
    const _Float16* wrow = Wp + (size_t)(ot*16 + col)*KPAD;
    for (int kc = 0; kc < KPAD; kc += 32) {
      v8h r1 = *reinterpret_cast<const v8h*>(wrow + kc + 8*h);
      v8h r2 = *reinterpret_cast<const v8h*>(wrow + kc + 16 + 8*h);
      v16h a = __builtin_shufflevector(r1, r2, 0,1,2,3,4,5,6,7,8,9,10,11,12,13,14,15);
      v16h bm = *reinterpret_cast<const v16h*>(sfB + (kc >> 5)*512 + lane*16);
      acc = __builtin_amdgcn_wmma_f32_16x16x32_f16(false, a, false, bm,
                                                   (short)0, acc, false, false);
    }
    if (pass == 0) {
      v8f s1 = acc, s2;
      #pragma unroll
      for (int r = 0; r < 8; ++r) s2[r] = acc[r]*acc[r];
      #pragma unroll
      for (int off = 1; off < 16; off <<= 1) {
        #pragma unroll
        for (int r = 0; r < 8; ++r) {
          s1[r] += __shfl_xor(s1[r], off, 32);
          s2[r] += __shfl_xor(s2[r], off, 32);
        }
      }
      if (col == 0) {
        #pragma unroll
        for (int r = 0; r < 8; ++r) {
          int ch = ot*16 + r + 8*h;
          atomicAdd(&sum[ch],   s1[r]);
          atomicAdd(&sumsq[ch], s2[r]);
        }
      }
    } else {
      int s = s0 + col;
      int n = s / KNN;
      #pragma unroll
      for (int r = 0; r < 8; ++r) {
        int ch = ot*16 + r + 8*h;
        float yh = acc[r]*Asc[ch] + Bsh[ch];
        yh = yh >= 0.f ? yh : SLOPE*yh;
        atomicMax(&xout[((size_t)b*C_out + ch)*NPTS + n], yh);
      }
    }
  }
}

__global__ void k_bn_finalize(const float* sum, const float* sumsq,
                              const float* g, const float* bb,
                              float* Asc, float* Bsh, int C_out) {
  int c = blockIdx.x*blockDim.x + threadIdx.x;
  if (c >= C_out) return;
  const float cnt = (float)(BATCH*NPTS*KNN);
  float mean = sum[c]/cnt;
  float var  = sumsq[c]/cnt - mean*mean;
  float A    = g[c] * rsqrtf(var + EPSV);
  Asc[c] = A;
  Bsh[c] = bb[c] - mean*A;
}

// fc1: (16 x 262144) @ (262144 x 512); float4 loads, in-register fp32->f16,
// split-K across blocks + fp32 atomicAdd of partials.
__global__ void k_fc1(const float* __restrict__ flat, const float* __restrict__ Wf,
                      float* __restrict__ out) {
  int ntile = blockIdx.x & 31;
  int kseg  = blockIdx.x >> 5;
  int wave = threadIdx.x >> 5, lane = threadIdx.x & 31;
  int col = lane & 15, h = lane >> 4;
  const int kper = FC1_K / KSPLIT;
  int kbeg = kseg * kper;
  const float* arow = flat + (size_t)col*FC1_K;
  const float* brow = Wf + (size_t)(ntile*16 + col)*FC1_K;
  v8f acc = {};
  for (int kc = kbeg + wave*32; kc < kbeg + kper; kc += 8*32) {
    float4 a0 = *reinterpret_cast<const float4*>(arow + kc + 8*h);
    float4 a1 = *reinterpret_cast<const float4*>(arow + kc + 8*h + 4);
    float4 a2 = *reinterpret_cast<const float4*>(arow + kc + 16 + 8*h);
    float4 a3 = *reinterpret_cast<const float4*>(arow + kc + 16 + 8*h + 4);
    float4 b0 = *reinterpret_cast<const float4*>(brow + kc + 8*h);
    float4 b1 = *reinterpret_cast<const float4*>(brow + kc + 8*h + 4);
    float4 b2 = *reinterpret_cast<const float4*>(brow + kc + 16 + 8*h);
    float4 b3 = *reinterpret_cast<const float4*>(brow + kc + 16 + 8*h + 4);
    v16h a, bm;
    a[0]=(_Float16)a0.x; a[1]=(_Float16)a0.y; a[2]=(_Float16)a0.z; a[3]=(_Float16)a0.w;
    a[4]=(_Float16)a1.x; a[5]=(_Float16)a1.y; a[6]=(_Float16)a1.z; a[7]=(_Float16)a1.w;
    a[8]=(_Float16)a2.x; a[9]=(_Float16)a2.y; a[10]=(_Float16)a2.z; a[11]=(_Float16)a2.w;
    a[12]=(_Float16)a3.x; a[13]=(_Float16)a3.y; a[14]=(_Float16)a3.z; a[15]=(_Float16)a3.w;
    bm[0]=(_Float16)b0.x; bm[1]=(_Float16)b0.y; bm[2]=(_Float16)b0.z; bm[3]=(_Float16)b0.w;
    bm[4]=(_Float16)b1.x; bm[5]=(_Float16)b1.y; bm[6]=(_Float16)b1.z; bm[7]=(_Float16)b1.w;
    bm[8]=(_Float16)b2.x; bm[9]=(_Float16)b2.y; bm[10]=(_Float16)b2.z; bm[11]=(_Float16)b2.w;
    bm[12]=(_Float16)b3.x; bm[13]=(_Float16)b3.y; bm[14]=(_Float16)b3.z; bm[15]=(_Float16)b3.w;
    acc = __builtin_amdgcn_wmma_f32_16x16x32_f16(false, a, false, bm,
                                                 (short)0, acc, false, false);
  }
  #pragma unroll
  for (int r = 0; r < 8; ++r) {
    int m = r + 8*h;                   // batch row
    atomicAdd(&out[m*FC1_N + ntile*16 + col], acc[r]);
  }
}

__global__ void k_bias_relu(const float* in, const float* bias, float* out, int M, int N) {
  int i = blockIdx.x*blockDim.x + threadIdx.x;
  if (i >= M*N) return;
  float v = in[i] + bias[i % N];
  out[i] = v > 0.f ? v : 0.f;
}

__global__ void k_fc_naive(const float* in, const float* W, const float* bias,
                           float* out, int M, int N, int Kd, int relu) {
  int i = blockIdx.x*blockDim.x + threadIdx.x;
  if (i >= M*N) return;
  int m = i / N, o = i % N;
  float s = bias[o];
  for (int j = 0; j < Kd; ++j) s += in[m*Kd + j] * W[o*Kd + j];
  if (relu) s = s > 0.f ? s : 0.f;
  out[i] = s;
}

extern "C" void kernel_launch(void* const* d_in, const int* in_sizes, int n_in,
                              void* d_out, int out_size, void* d_ws, size_t ws_size,
                              hipStream_t stream) {
  (void)in_sizes; (void)n_in; (void)out_size; (void)ws_size;
  const float* x_raw   = (const float*)d_in[0];
  const float* convW[4]= {(const float*)d_in[1], (const float*)d_in[4],
                          (const float*)d_in[7], (const float*)d_in[10]};
  const float* bnG[4]  = {(const float*)d_in[2], (const float*)d_in[5],
                          (const float*)d_in[8], (const float*)d_in[11]};
  const float* bnB[4]  = {(const float*)d_in[3], (const float*)d_in[6],
                          (const float*)d_in[9], (const float*)d_in[12]};
  const float* fc1_w = (const float*)d_in[13]; const float* fc1_b = (const float*)d_in[14];
  const float* fc2_w = (const float*)d_in[15]; const float* fc2_b = (const float*)d_in[16];
  const float* fc3_w = (const float*)d_in[17]; const float* fc3_b = (const float*)d_in[18];

  char* ws = (char*)d_ws;
  float* act0  = (float*)(ws);                              // 16 MB
  float* act1  = (float*)(ws + (size_t)16*1024*1024);       // 16 MB
  float* xx    = (float*)(ws + (size_t)32*1024*1024);       // 64 KB
  int*   idx   = (int*)  (ws + (size_t)32*1024*1024 + 256*1024); // 1.25 MB
  float* sum   = (float*)(ws + (size_t)34*1024*1024);
  float* sumsq = sum + 256;
  float* Asc   = sum + 512;
  float* Bsh   = sum + 768;
  float* fcbuf0= sum + 1024;
  float* fcbuf1= fcbuf0 + 16*512;
  _Float16* wp_base = (_Float16*)(ws + (size_t)35*1024*1024);

  int Cin[4]  = {3, 64, 64, 128};
  int Cout[4] = {64, 64, 128, 256};
  int KPAD[4] = {32, 128, 128, 256};
  _Float16* Wp[4];
  {
    size_t off = 0;
    for (int L = 0; L < 4; ++L) { Wp[L] = wp_base + off; off += (size_t)Cout[L]*KPAD[L]; }
  }
  float* bufs[2] = {act0, act1};

  k_transpose_x<<<(BATCH*NPTS*3 + 255)/256, 256, 0, stream>>>(x_raw, act0);
  for (int L = 0; L < 4; ++L) {
    int n = Cout[L]*KPAD[L];
    k_prep_w<<<(n + 255)/256, 256, 0, stream>>>(convW[L], Wp[L], Cout[L], 2*Cin[L], KPAD[L]);
  }

  int cur = 0;
  for (int L = 0; L < 4; ++L) {
    float* xin  = bufs[cur];
    float* xout = bufs[cur ^ 1];
    k_sqnorm<<<(BATCH*NPTS + 255)/256, 256, 0, stream>>>(xin, xx, Cin[L]);
    k_knn<<<BATCH*NPTS/8, 256, 0, stream>>>(xin, xx, idx, Cin[L]);
    k_fill<<<2, 256, 0, stream>>>(sum, 0.f, 512);           // sum + sumsq
    int blocks = BATCH * (NPTS*KNN/16);
    k_edgeconv<<<blocks, 256, 0, stream>>>(xin, idx, Wp[L], sum, sumsq,
                                           Asc, Bsh, xout, Cin[L], Cout[L], KPAD[L], 0);
    k_bn_finalize<<<1, 256, 0, stream>>>(sum, sumsq, bnG[L], bnB[L], Asc, Bsh, Cout[L]);
    int outn = BATCH*Cout[L]*NPTS;
    k_fill<<<(outn + 255)/256, 256, 0, stream>>>(xout, -3.4e38f, outn);
    k_edgeconv<<<blocks, 256, 0, stream>>>(xin, idx, Wp[L], sum, sumsq,
                                           Asc, Bsh, xout, Cin[L], Cout[L], KPAD[L], 1);
    cur ^= 1;
  }
  float* x4 = bufs[cur];   // (B,256,N) contiguous == flat(B, 262144)

  k_fill<<<(16*512 + 255)/256, 256, 0, stream>>>(fcbuf0, 0.f, 16*512);
  k_fc1<<<32*KSPLIT, 256, 0, stream>>>(x4, fc1_w, fcbuf0);
  k_bias_relu<<<(16*512 + 255)/256, 256, 0, stream>>>(fcbuf0, fc1_b, fcbuf1, 16, 512);
  k_fc_naive<<<(16*512 + 255)/256, 256, 0, stream>>>(fcbuf1, fc2_w, fc2_b, fcbuf0, 16, 512, 512, 1);
  k_fc_naive<<<1, 32, 0, stream>>>(fcbuf0, fc3_w, fc3_b, (float*)d_out, 16, 2, 512, 0);
}
